// Bool_40793599377512
// MI455X (gfx1250) — matrix operand order repaired and checked by reading, not compiled, not run on hardware
//
#include <hip/hip_runtime.h>

#define NROWS   8192
#define DIN     4096
#define DOUT    4096
#define KT      32
#define LDT     40          // padded LDS row stride in halves (80B, 16B aligned)
#define PERM_PAD 8320       // NROWS + 128 (gap padding for expert-1 segment)
#define BM      128
#define BN      256
#define NSTEPS  (DIN / KT)  // 128

typedef __attribute__((ext_vector_type(16))) _Float16 v16h;
typedef __attribute__((ext_vector_type(8)))  _Float16 v8h;
typedef __attribute__((ext_vector_type(8)))  float    v8f;

// ---------------- Kernel 1: routing g[n] = (sum(x[n]) > 0) ----------------
__global__ void route_kernel(const float* __restrict__ x, int* __restrict__ g) {
    __shared__ float red[256];
    const int row = blockIdx.x;
    const float* xr = x + (size_t)row * DIN;
    float s = 0.f;
    for (int i = threadIdx.x; i < DIN; i += 256) s += xr[i];
    red[threadIdx.x] = s;
    __syncthreads();
    for (int off = 128; off > 0; off >>= 1) {
        if (threadIdx.x < off) red[threadIdx.x] += red[threadIdx.x + off];
        __syncthreads();
    }
    if (threadIdx.x == 0) g[row] = (red[0] > 0.f) ? 1 : 0;
}

// ------------- Kernel 2: stable partition of rows by expert ---------------
__global__ void partition_kernel(const int* __restrict__ g, int* __restrict__ perm,
                                 int* __restrict__ meta) {
    __shared__ int scan[1024];
    __shared__ int off0s, off1s;
    const int tid = threadIdx.x;

    for (int i = tid; i < PERM_PAD; i += 1024) perm[i] = -1;

    int c1l = 0;
    for (int i = tid; i < NROWS; i += 1024) c1l += g[i];
    scan[tid] = c1l;
    __syncthreads();
    for (int off = 512; off > 0; off >>= 1) {
        if (tid < off) scan[tid] += scan[tid + off];
        __syncthreads();
    }
    const int c1 = scan[0];
    const int c0 = NROWS - c1;
    const int B  = ((c0 + 127) >> 7) << 7;
    if (tid == 0) { meta[0] = c0; meta[1] = B; off0s = 0; off1s = 0; }
    __syncthreads();

    for (int chunk = 0; chunk < NROWS; chunk += 1024) {
        const int idx = chunk + tid;
        const int gv = g[idx];
        const int packed = gv ? (1 << 16) : 1;
        scan[tid] = packed;
        __syncthreads();
        for (int off = 1; off < 1024; off <<= 1) {
            int add = (tid >= off) ? scan[tid - off] : 0;
            __syncthreads();
            scan[tid] += add;
            __syncthreads();
        }
        const int inc   = scan[tid];
        const int tot   = scan[1023];
        const int excl0 = (inc & 0xFFFF) - (packed & 0xFFFF);
        const int excl1 = (inc >> 16)    - (packed >> 16);
        if (gv == 0) perm[off0s + excl0]     = idx;
        else         perm[B + off1s + excl1] = idx;
        __syncthreads();
        if (tid == 0) { off0s += tot & 0xFFFF; off1s += tot >> 16; }
        __syncthreads();
    }
}

// ---------------- WMMA fragment loaders (per ISA VGPR layouts) ------------
__device__ __forceinline__ v16h load_a_frag(const _Float16* As, int rowBase, int lane) {
    const int m = lane & 15, kg = lane >> 4;
    const _Float16* pr = As + (rowBase + m) * LDT + kg * 8;
    v8h lo = *(const v8h*)pr;          // K = 8*kg .. 8*kg+7
    v8h hi = *(const v8h*)(pr + 16);   // K = 16+8*kg ..
    v16h r;
#pragma unroll
    for (int i = 0; i < 8; ++i) { r[i] = lo[i]; r[i + 8] = hi[i]; }
    return r;
}

__device__ __forceinline__ v16h load_b_frag(const _Float16* Bs, int colBase, int lane) {
    const int n = lane & 15, kg = lane >> 4;
    const _Float16* pr = Bs + (colBase + n) * LDT + kg * 8;
    v8h lo = *(const v8h*)pr;
    v8h hi = *(const v8h*)(pr + 16);
    v16h r;
#pragma unroll
    for (int i = 0; i < 8; ++i) { r[i] = lo[i]; r[i + 8] = hi[i]; }
    return r;
}

// ---------------- staging helpers -----------------------------------------
__device__ __forceinline__ void stage_a(const float* aptr, _Float16* dst, int kk) {
    float4 a0{}, a1{}, a2{}, a3{};
    if (aptr) {
        const float4* ap = (const float4*)(aptr + kk);
        a0 = ap[0]; a1 = ap[1]; a2 = ap[2]; a3 = ap[3];
    }
    v8h h0 = { (_Float16)a0.x,(_Float16)a0.y,(_Float16)a0.z,(_Float16)a0.w,
               (_Float16)a1.x,(_Float16)a1.y,(_Float16)a1.z,(_Float16)a1.w };
    v8h h1 = { (_Float16)a2.x,(_Float16)a2.y,(_Float16)a2.z,(_Float16)a2.w,
               (_Float16)a3.x,(_Float16)a3.y,(_Float16)a3.z,(_Float16)a3.w };
    *(v8h*)dst       = h0;
    *(v8h*)(dst + 8) = h1;
}

__device__ __forceinline__ void stage_b16(const float* gp, _Float16* Bbuf,
                                          int n0, int bk, int rot) {
    const float4* bp = (const float4*)gp;
    float4 b0 = bp[0], b1 = bp[1], b2 = bp[2], b3 = bp[3];
    float bvv[16] = { b0.x,b0.y,b0.z,b0.w, b1.x,b1.y,b1.z,b1.w,
                      b2.x,b2.y,b2.z,b2.w, b3.x,b3.y,b3.z,b3.w };
#pragma unroll
    for (int u = 0; u < 16; ++u) {
        const int uu = (u + rot) & 15;     // lane-rotated to spread LDS banks
        Bbuf[(n0 + uu) * LDT + bk] = (_Float16)bvv[uu];
    }
}

// exact tanh via e^{-2|x|}: 2 trans + ~5 valu, ~1e-7 rel err
__device__ __forceinline__ float fast_tanh(float v) {
    const float ax = __builtin_fabsf(v);
    const float z  = __builtin_amdgcn_exp2f(ax * -2.885390081777927f); // exp(-2ax)
    const float t  = (1.f - z) * __builtin_amdgcn_rcpf(1.f + z);
    return __builtin_copysignf(t, v);
}

// ---------------- Kernel 3: gathered single-expert tiled GEMM -------------
// Block tile 128(M) x 256(N), K-step 32, double-buffered LDS.
// 8 waves in 2(M) x 4(N) grid -> 64x64 per wave = 16 WMMAs / K-step.
__global__ __launch_bounds__(256)
void moe_gemm_kernel(const float* __restrict__ x, const float* __restrict__ W,
                     const float* __restrict__ bias, const int* __restrict__ perm,
                     const int* __restrict__ meta, float* __restrict__ out) {
    __shared__ _Float16 As[2][BM * LDT];
    __shared__ _Float16 Bs[2][BN * LDT];
    __shared__ int rowIdx[BM];

    const int c0   = meta[0];
    const int Bseg = meta[1];
    const int span = Bseg + (NROWS - c0);
    const int p    = blockIdx.y * BM;
    if (p >= span) return;
    const int e    = (p >= Bseg) ? 1 : 0;
    const int col0 = blockIdx.x * BN;

    const int tid = threadIdx.x;
    if (tid < BM) rowIdx[tid] = perm[p + tid];
    __syncthreads();

    const int lane   = tid & 31;
    const int wid    = tid >> 5;
    const int rowOff = (wid & 1) * 64;     // wave's 64 output rows
    const int colOff = (wid >> 1) * 64;    // wave's 64 output cols

    v8f acc[4][4];
#pragma unroll
    for (int i = 0; i < 4; ++i)
#pragma unroll
        for (int j = 0; j < 4; ++j) acc[i][j] = (v8f){0,0,0,0,0,0,0,0};

    // A-stage: 2 threads per row, 16 floats each
    const int ai = tid >> 1;
    const int ah = tid & 1;
    const int arow = rowIdx[ai];
    const float* aptr = (arow >= 0) ? (x + (size_t)arow * DIN + ah * 16) : nullptr;
    const int aslot = ai * LDT + ah * 16;

    // B-stage: k = tid>>3, two 16-col chunks at (tid&7)*16 and +128
    const int bk = tid >> 3;
    const int bj = tid & 7;
    const float* bptr = W + (size_t)e * DIN * DOUT + (size_t)bk * DOUT + col0 + bj * 16;

    // prologue: fill buffer 0
    stage_a(aptr, &As[0][aslot], 0);
    stage_b16(bptr,       &Bs[0][0], bj * 16,       bk, 2 * bj);
    stage_b16(bptr + 128, &Bs[0][0], bj * 16 + 128, bk, 2 * bj);
    __syncthreads();

    for (int s = 0; s < NSTEPS; ++s) {
        const int cur = s & 1;
        const int nxt = cur ^ 1;

        // stage next K-tile into the other buffer (overlaps with WMMA below)
        if (s + 1 < NSTEPS) {
            const int kk = (s + 1) * KT;
            stage_a(aptr, &As[nxt][aslot], kk);
            const float* gb = bptr + (size_t)kk * DOUT;
            stage_b16(gb,       &Bs[nxt][0], bj * 16,       bk, 2 * bj);
            stage_b16(gb + 128, &Bs[nxt][0], bj * 16 + 128, bk, 2 * bj);
            if (s + 2 < NSTEPS) {
                if (aptr) __builtin_prefetch(aptr + (s + 2) * KT, 0, 1);
                __builtin_prefetch(bptr + (size_t)(s + 2) * KT * DOUT, 0, 1);
            }
        }

        // 16 WMMAs per wave from current buffer
        v16h bF[4];
#pragma unroll
        for (int cb = 0; cb < 4; ++cb)
            bF[cb] = load_b_frag(&Bs[cur][0], colOff + cb * 16, lane);
#pragma unroll
        for (int rb = 0; rb < 4; ++rb) {
            v16h aF = load_a_frag(&As[cur][0], rowOff + rb * 16, lane);
#pragma unroll
            for (int cb = 0; cb < 4; ++cb)
                acc[rb][cb] = __builtin_amdgcn_wmma_f32_16x16x32_f16(
                    false, aF, false, bF[cb], (short)0, acc[rb][cb], false, false);
        }
        __syncthreads();
    }

    // ---- epilogue: bias + tanh, scatter rows back ----
    const int n  = lane & 15;
    const int kg = lane >> 4;
#pragma unroll
    for (int cb = 0; cb < 4; ++cb) {
        const int col = col0 + colOff + cb * 16 + n;
        const float bv = bias[e * DOUT + col];
#pragma unroll
        for (int rb = 0; rb < 4; ++rb) {
#pragma unroll
            for (int r = 0; r < 8; ++r) {
                const int lm  = rowOff + rb * 16 + r + 8 * kg;   // C: M = r + 8*kg
                const int row = rowIdx[lm];
                if (row >= 0)
                    out[(size_t)row * DOUT + col] = fast_tanh(acc[rb][cb][r] + bv);
            }
        }
    }
}

// ------------------------------- launcher ---------------------------------
extern "C" void kernel_launch(void* const* d_in, const int* in_sizes, int n_in,
                              void* d_out, int out_size, void* d_ws, size_t ws_size,
                              hipStream_t stream) {
    const float* x = (const float*)d_in[0];
    const float* W = (const float*)d_in[1];
    const float* b = (const float*)d_in[2];
    float* out = (float*)d_out;

    int* g    = (int*)d_ws;          // NROWS ints
    int* perm = g + NROWS;           // PERM_PAD ints
    int* meta = perm + PERM_PAD;     // 2 ints

    route_kernel<<<NROWS, 256, 0, stream>>>(x, g);
    partition_kernel<<<1, 1024, 0, stream>>>(g, perm, meta);
    dim3 grid(DOUT / BN, PERM_PAD / BM);   // (16, 65)
    moe_gemm_kernel<<<grid, 256, 0, stream>>>(x, W, b, perm, meta, out);
}
